// MultiHeadAttentionLayer_31164282699913
// MI455X (gfx1250) — matrix operand order repaired and checked by reading, not compiled, not run on hardware
//
#include <hip/hip_runtime.h>

typedef _Float16 v16h __attribute__((ext_vector_type(16)));
typedef _Float16 v8h  __attribute__((ext_vector_type(8)));
typedef float    v8f  __attribute__((ext_vector_type(8)));
typedef float    v4f  __attribute__((ext_vector_type(4)));

#define IN_DIM    1024
#define HEAD_DIM  64
#define NUM_HEADS 16
#define SEQ       4096
#define QKV_ELEMS ((size_t)NUM_HEADS * SEQ * HEAD_DIM)  // 4,194,304 halves per buffer

static __device__ __forceinline__ v16h hcat(v8h lo, v8h hi) {
  return __builtin_shufflevector(lo, hi, 0, 1, 2, 3, 4, 5, 6, 7, 8, 9, 10, 11, 12, 13, 14, 15);
}

static __device__ __forceinline__ v8f wmma_f16(v16h a, v16h b, v8f c) {
  // D = A(16x32 f16) x B(32x16 f16) + C(16x16 f32)
  return __builtin_amdgcn_wmma_f32_16x16x32_f16(false, a, false, b, (short)0, c, false, false);
}

static __device__ __forceinline__ v8h cvt8(v4f a, v4f b) {
  v8h r = {(_Float16)a[0], (_Float16)a[1], (_Float16)a[2], (_Float16)a[3],
           (_Float16)b[0], (_Float16)b[1], (_Float16)b[2], (_Float16)b[3]};
  return r;
}

// Async global->LDS copy (CDNA5 path): 16B per lane, tracked with ASYNCcnt.
static __device__ __forceinline__ void async_load_b128(void* lds_ptr, const void* gptr) {
  unsigned lds_off = (unsigned)(size_t)lds_ptr;  // generic addr low 32b = LDS offset
  unsigned long long ga = (unsigned long long)(size_t)gptr;
  asm volatile("global_load_async_to_lds_b128 %0, %1, off" ::"v"(lds_off), "v"(ga) : "memory");
}
static __device__ __forceinline__ void wait_async0() {
  asm volatile("s_wait_asynccnt 0" ::: "memory");
}

// ---------------------------------------------------------------------------
// Kernel 1: fused QKV projection.  Y = h @ W + b, converted to f16.
//   z==0: Q, scaled by 1/sqrt(64)=0.125, layout [H][S][D]
//   z==1: K, layout [H][S][D]
//   z==2: V, layout [H][D][S]  (transposed: feeds PV A-fragments directly)
// Block tile M=128 x N=64, K-tile 64.  Global loads of tile k+1 are issued
// before tile k's WMMAs (software pipeline over the VGPR bounce needed for
// the f32->f16 conversion).
// ---------------------------------------------------------------------------
__global__ __launch_bounds__(256)
void qkv_proj_kernel(const float* __restrict__ hin,
                     const float* __restrict__ Wq, const float* __restrict__ Wk,
                     const float* __restrict__ Wv,
                     const float* __restrict__ bq, const float* __restrict__ bk,
                     const float* __restrict__ bv,
                     _Float16* __restrict__ q16, _Float16* __restrict__ k16,
                     _Float16* __restrict__ vt16) {
  __shared__ _Float16 Ah[128][72];  // h tile [m][k], row stride 144B (16B-aligned)
  __shared__ _Float16 Bw[64][72];   // W tile transposed [n][k]

  const int z = blockIdx.z;
  const float* W    = (z == 0) ? Wq : (z == 1) ? Wk : Wv;
  const float* bias = (z == 0) ? bq : (z == 1) ? bk : bv;
  _Float16* outp    = (z == 0) ? q16 : (z == 1) ? k16 : vt16;

  const int m0 = blockIdx.x * 128;
  const int n0 = blockIdx.y * 64;

  const int tid  = threadIdx.x;
  const int wave = tid >> 5;
  const int lane = tid & 31;
  const int lhi  = lane >> 4;
  const int llo  = lane & 15;

  // staging maps
  const int arow = tid >> 1;        // 0..127
  const int akh  = (tid & 1) * 32;  // k offset 0 or 32
  const int bkr  = tid >> 2;        // 0..63 (k row of W)
  const int bn16 = (tid & 3) * 16;  // 0..48

  const float* asrc = hin + (size_t)(m0 + arow) * IN_DIM + akh;
  const float* bsrc = W + (size_t)bkr * IN_DIM + n0 + bn16;

  v8f acc[4] = {{}, {}, {}, {}};
  v4f ar[8], br[4];

  // prologue: fetch tile k0=0 into VGPRs
#pragma unroll
  for (int i = 0; i < 8; ++i) ar[i] = *(const v4f*)(asrc + i * 4);
#pragma unroll
  for (int i = 0; i < 4; ++i) br[i] = *(const v4f*)(bsrc + i * 4);

  for (int k0 = 0; k0 < IN_DIM; k0 += 64) {
    __syncthreads();  // previous tile's LDS reads complete
    // --- commit staged VGPRs to LDS (f32 -> f16) ---
#pragma unroll
    for (int i = 0; i < 4; ++i)
      *(v8h*)&Ah[arow][akh + i * 8] = cvt8(ar[2 * i], ar[2 * i + 1]);
#pragma unroll
    for (int i = 0; i < 4; ++i)
#pragma unroll
      for (int j = 0; j < 4; ++j) Bw[bn16 + i * 4 + j][bkr] = (_Float16)br[i][j];
    __syncthreads();

    // --- prefetch next tile's globals under this tile's WMMAs ---
    if (k0 + 64 < IN_DIM) {
#pragma unroll
      for (int i = 0; i < 8; ++i) ar[i] = *(const v4f*)(asrc + (k0 + 64) + i * 4);
#pragma unroll
      for (int i = 0; i < 4; ++i) br[i] = *(const v4f*)(bsrc + (size_t)(k0 + 64) * IN_DIM + i * 4);
    }

    const int am = wave * 16 + llo;
#pragma unroll
    for (int ks = 0; ks < 2; ++ks) {
      // batch all fragment loads, then issue the WMMA group
      v16h a = hcat(*(const v8h*)&Ah[am][ks * 32 + lhi * 8],
                    *(const v8h*)&Ah[am][ks * 32 + 16 + lhi * 8]);
      v16h b[4];
#pragma unroll
      for (int j = 0; j < 4; ++j) {
        const int bn = j * 16 + llo;  // B lane = output column n, K at g*8 and 16+g*8
        b[j] = hcat(*(const v8h*)&Bw[bn][ks * 32 + lhi * 8],
                    *(const v8h*)&Bw[bn][ks * 32 + 16 + lhi * 8]);
      }
#pragma unroll
      for (int j = 0; j < 4; ++j) acc[j] = wmma_f16(a, b[j], acc[j]);
    }
  }

  // --- epilogue: +bias, (Q only) *0.125, convert to f16, scatter per layout ---
  const float scl = (z == 0) ? 0.125f : 1.0f;
#pragma unroll
  for (int j = 0; j < 4; ++j) {
    const int n = n0 + j * 16 + llo;
    const int head = n >> 6;
    const int d = n & 63;
    const float bn = bias[n];
#pragma unroll
    for (int r = 0; r < 8; ++r) {
      const int s = m0 + wave * 16 + r + 8 * lhi;  // C layout: M = r + 8*lane_half
      const float v = (acc[j][r] + bn) * scl;
      if (z == 2)
        outp[(size_t)head * (SEQ * HEAD_DIM) + (size_t)d * SEQ + s] = (_Float16)v;
      else
        outp[(size_t)head * (SEQ * HEAD_DIM) + (size_t)s * HEAD_DIM + d] = (_Float16)v;
    }
  }
}

// ---------------------------------------------------------------------------
// Kernel 2: flash attention, transposed-score formulation.
// One block = 1 head x 128 query rows (8 waves x 16 queries).
// K/V chunks double-buffered in LDS via global_load_async_to_lds_b128:
// chunk i+1's copy is in flight (ASYNCcnt) while chunk i is computed; one
// barrier + one s_wait_asynccnt per 32-key chunk.
// Per chunk per wave:
//   S^T = K . Q^T      (A = K tile, B = Q^T)  -> lane owns one query column
//   per-lane online softmax (2 xor-16 shuffles total)
//   O^T += V^T . P     (A = V^T tile, B = P assembled via xor-16 exchanges)
// ---------------------------------------------------------------------------
__global__ __launch_bounds__(256)
void flash_attn_kernel(const _Float16* __restrict__ q16,
                       const _Float16* __restrict__ k16,
                       const _Float16* __restrict__ vt16,
                       float* __restrict__ out) {
  __shared__ _Float16 Kl[2][32][72];  // K chunk [t][d], double-buffered
  __shared__ _Float16 Vl[2][64][40];  // V chunk transposed [d][t], double-buffered

  const int head = blockIdx.y;
  const int s0b  = blockIdx.x * 128;
  const int tid  = threadIdx.x;
  const int wave = tid >> 5;
  const int lane = tid & 31;
  const int lhi  = lane >> 4;
  const int llo  = lane & 15;
  const int sw   = s0b + wave * 16;

  // Q^T B-fragments (scale folded in): lane n = query llo, elements = d = g*8+e, 16+g*8+e
  const _Float16* qrow = q16 + ((size_t)head * SEQ + sw + llo) * HEAD_DIM;
  const v16h qb0 = hcat(*(const v8h*)(qrow + lhi * 8), *(const v8h*)(qrow + 16 + lhi * 8));
  const v16h qb1 = hcat(*(const v8h*)(qrow + 32 + lhi * 8), *(const v8h*)(qrow + 48 + lhi * 8));

  v8f o[4] = {{}, {}, {}, {}};       // O^T: tile j covers d = j*16 + (r + 8*lhi)
  float mrun = -1e30f, lrun = 0.0f;  // per-lane stats for query (sw + llo)

  const _Float16* kbase = k16 + (size_t)head * SEQ * HEAD_DIM;
  const _Float16* vbase = vt16 + (size_t)head * SEQ * HEAD_DIM;

  // staging maps: K = 32 t x 64 d halves, V = 64 d x 32 t halves, 16B per lane
  const int ktr = tid >> 3, kd8 = (tid & 7) * 8;
  const int vdr = tid >> 2, vt8 = (tid & 3) * 8;

  auto issue_chunk = [&](int t0, int buf) {
    async_load_b128(&Kl[buf][ktr][kd8], kbase + (size_t)(t0 + ktr) * HEAD_DIM + kd8);
    async_load_b128(&Vl[buf][vdr][vt8], vbase + (size_t)vdr * SEQ + t0 + vt8);
  };

  constexpr int NCHUNK = SEQ / 32;
  issue_chunk(0, 0);

  for (int i = 0; i < NCHUNK; ++i) {
    const int cur = i & 1;
    wait_async0();     // own async writes (chunk i) landed
    __syncthreads();   // ... and everyone else's; also fences reads of buf cur^1
    if (i + 1 < NCHUNK) issue_chunk((i + 1) * 32, cur ^ 1);  // overlap next copy

    // --- S^T tiles: batch all K A-fragments, then the WMMA group ---
    v16h ka[4];
    ka[0] = hcat(*(const v8h*)&Kl[cur][llo][lhi * 8],
                 *(const v8h*)&Kl[cur][llo][16 + lhi * 8]);
    ka[1] = hcat(*(const v8h*)&Kl[cur][llo][32 + lhi * 8],
                 *(const v8h*)&Kl[cur][llo][48 + lhi * 8]);
    ka[2] = hcat(*(const v8h*)&Kl[cur][16 + llo][lhi * 8],
                 *(const v8h*)&Kl[cur][16 + llo][16 + lhi * 8]);
    ka[3] = hcat(*(const v8h*)&Kl[cur][16 + llo][32 + lhi * 8],
                 *(const v8h*)&Kl[cur][16 + llo][48 + lhi * 8]);
    v8f c0 = {}, c1 = {};
    c0 = wmma_f16(ka[0], qb0, c0);
    c0 = wmma_f16(ka[1], qb1, c0);
    c1 = wmma_f16(ka[2], qb0, c1);
    c1 = wmma_f16(ka[3], qb1, c1);

    // --- V^T fragments issued now so the ds_loads fly during the softmax ---
    v16h va[4];
#pragma unroll
    for (int j = 0; j < 4; ++j)
      va[j] = hcat(*(const v8h*)&Vl[cur][j * 16 + llo][lhi * 16],
                   *(const v8h*)&Vl[cur][j * 16 + llo][lhi * 16 + 8]);

    // --- per-lane online softmax for this lane's query column ---
    float lm = fmaxf(c0[0], c1[0]);
#pragma unroll
    for (int r = 1; r < 8; ++r) lm = fmaxf(lm, fmaxf(c0[r], c1[r]));
    lm = fmaxf(lm, __shfl_xor(lm, 16, 32));  // merge the two t-halves
    const float mnew  = fmaxf(mrun, lm);
    const float alpha = __expf(mrun - mnew);
    float p0[8], p1[8];
    float ls = 0.0f;
#pragma unroll
    for (int r = 0; r < 8; ++r) {
      p0[r] = __expf(c0[r] - mnew);
      p1[r] = __expf(c1[r] - mnew);
      ls += p0[r] + p1[r];
    }
    ls += __shfl_xor(ls, 16, 32);
    lrun = lrun * alpha + ls;
    mrun = mnew;
#pragma unroll
    for (int j = 0; j < 4; ++j)
#pragma unroll
      for (int r = 0; r < 8; ++r) o[j][r] *= alpha;  // alpha uniform per lane

    // --- assemble P as B-fragment (lane n = query, elements e -> t = g*16+e) ---
    // own half supplies 8 t-values, partner (xor-16) supplies the other 8.
    float x0[8], x1[8];
#pragma unroll
    for (int r = 0; r < 8; ++r) {
      x0[r] = __shfl_xor(p0[r], 16, 32);
      x1[r] = __shfl_xor(p1[r], 16, 32);
    }
    v16h pb;
#pragma unroll
    for (int e = 0; e < 8; ++e) {
      pb[e]     = (_Float16)(lhi ? x1[e] : p0[e]);  // t = g*16 + e
      pb[8 + e] = (_Float16)(lhi ? p1[e] : x0[e]);  // t = g*16 + 8 + e
    }

    // --- O^T += V^T . P ---
#pragma unroll
    for (int j = 0; j < 4; ++j) o[j] = wmma_f16(va[j], pb, o[j]);
  }

  // --- epilogue: each lane owns one output row; contiguous v4f stores ---
  const float inv = 1.0f / lrun;
  float* orow = out + (size_t)(sw + llo) * (NUM_HEADS * HEAD_DIM) + head * HEAD_DIM;
#pragma unroll
  for (int j = 0; j < 4; ++j) {
    v4f lo = {o[j][0] * inv, o[j][1] * inv, o[j][2] * inv, o[j][3] * inv};
    v4f hi = {o[j][4] * inv, o[j][5] * inv, o[j][6] * inv, o[j][7] * inv};
    *(v4f*)(orow + j * 16 + 8 * lhi)     = lo;  // d = j*16 + 8*lhi + 0..3
    *(v4f*)(orow + j * 16 + 8 * lhi + 4) = hi;  // d = j*16 + 8*lhi + 4..7
  }
}

extern "C" void kernel_launch(void* const* d_in, const int* in_sizes, int n_in,
                              void* d_out, int out_size, void* d_ws, size_t ws_size,
                              hipStream_t stream) {
  (void)in_sizes; (void)n_in; (void)out_size; (void)ws_size;
  const float* h  = (const float*)d_in[0];
  const float* Wq = (const float*)d_in[1];
  const float* Wk = (const float*)d_in[2];
  const float* Wv = (const float*)d_in[3];
  const float* bq = (const float*)d_in[4];
  const float* bk = (const float*)d_in[5];
  const float* bv = (const float*)d_in[6];
  float* out = (float*)d_out;

  _Float16* ws   = (_Float16*)d_ws;  // needs 3 * 8 MB = 24 MB
  _Float16* q16  = ws;
  _Float16* k16  = ws + QKV_ELEMS;
  _Float16* vt16 = ws + 2 * QKV_ELEMS;

  dim3 gproj(SEQ / 128, (NUM_HEADS * HEAD_DIM) / 64, 3);
  qkv_proj_kernel<<<gproj, 256, 0, stream>>>(h, Wq, Wk, Wv, bq, bk, bv, q16, k16, vt16);

  dim3 gattn(SEQ / 128, NUM_HEADS);
  flash_attn_kernel<<<gattn, 256, 0, stream>>>(q16, k16, vt16, out);
}
